// CrossAttention1d_76149770158595
// MI455X (gfx1250) — matrix operand order repaired and checked by reading, not compiled, not run on hardware
//
#include <hip/hip_runtime.h>

#define N_ 4
#define C_ 512
#define L_ 2048
#define H_ 8
#define DH_ 64
#define EPS_ 1e-5f

typedef __attribute__((ext_vector_type(16))) _Float16 v16h;
typedef __attribute__((ext_vector_type(8)))  float    v8f;

union V16H { v16h v; uint4 u[2]; };
union H8   { uint4 u; _Float16 h[8]; };

// A-fragment (16x32 f16): lane holds row m=lane&15; element j -> k = (j&7) + 8*half + 16*(j>>3)
__device__ __forceinline__ v16h ld_a32(const _Float16* base, int half) {
    V16H r;
    r.u[0] = *reinterpret_cast<const uint4*>(base + 8 * half);
    r.u[1] = *reinterpret_cast<const uint4*>(base + 16 + 8 * half);
    return r.v;
}

// B-fragment (32x16 f16): lane holds col n=lane&15; element j -> k = (j&7) + 8*half + 16*(j>>3)
__device__ __forceinline__ v16h ld_b16(const _Float16* base, int half) {
    V16H r;
    r.u[0] = *reinterpret_cast<const uint4*>(base + 8 * half);
    r.u[1] = *reinterpret_cast<const uint4*>(base + 16 + 8 * half);
    return r.v;
}

__device__ __forceinline__ v8f wmma16(v16h a, v16h b, v8f c) {
    return __builtin_amdgcn_wmma_f32_16x16x32_f16(false, a, false, b, (short)0, c, false, false);
}

// xor-lane permute within the 16-lane row: VALU v_permlane16_b32 (no DS counter).
__device__ __forceinline__ float permx16(float v, unsigned lo, unsigned hi, int mask) {
#if __has_builtin(__builtin_amdgcn_permlane16)
    unsigned u = __builtin_bit_cast(unsigned, v);
    u = __builtin_amdgcn_permlane16(u, u, lo, hi, false, false);
    (void)mask;
    return __builtin_bit_cast(float, u);
#else
    (void)lo; (void)hi;
    return __shfl_xor(v, mask, 32);
#endif
}

// max across the 16-lane half-row (butterfly, 4 VALU cross-lane steps)
__device__ __forceinline__ float rowmax16(float v) {
    v = fmaxf(v, permx16(v, 0x67452301u, 0xEFCDAB89u, 1));
    v = fmaxf(v, permx16(v, 0x54761032u, 0xDCFE98BAu, 2));
    v = fmaxf(v, permx16(v, 0x32107654u, 0xBA98FEDCu, 4));
    v = fmaxf(v, permx16(v, 0xFEDCBA98u, 0x76543210u, 8));
    return v;
}

__device__ __forceinline__ float fast_rcp(float x) {
#if __has_builtin(__builtin_amdgcn_rcpf)
    return __builtin_amdgcn_rcpf(x);
#else
    return 1.0f / x;
#endif
}

// ---------------------------------------------------------------- stats
__global__ void zero_stats_kernel(float* stats) {
    if (threadIdx.x < 16) stats[threadIdx.x] = 0.0f;
}

// grid: (256 chunks, N, 2 tensors); 4096 elems / block
__global__ void stats_kernel(const float* __restrict__ a, const float* __restrict__ b,
                             float* __restrict__ stats) {
    const float* x = (blockIdx.z == 0) ? a : b;
    const int n = blockIdx.y;
    const float* xs = x + (size_t)n * C_ * L_;
    const int tid = threadIdx.x;
    const size_t start = (size_t)blockIdx.x * 4096 + tid;
    float s = 0.f, s2 = 0.f;
#pragma unroll
    for (int i = 0; i < 16; ++i) {
        float v = xs[start + (size_t)i * 256];
        s += v; s2 += v * v;
    }
    __shared__ float red0[256], red1[256];
    red0[tid] = s; red1[tid] = s2;
    __syncthreads();
    for (int off = 128; off > 0; off >>= 1) {
        if (tid < off) { red0[tid] += red0[tid + off]; red1[tid] += red1[tid + off]; }
        __syncthreads();
    }
    if (tid == 0) {
        atomicAdd(&stats[(blockIdx.z * N_ + n) * 2 + 0], red0[0]);
        atomicAdd(&stats[(blockIdx.z * N_ + n) * 2 + 1], red1[0]);
    }
}

// grid: NCL/1024 blocks of 256; 4 elems/thread
__global__ void normalize_kernel(const float* __restrict__ x, const float* __restrict__ w,
                                 const float* __restrict__ bb, const float* __restrict__ stats,
                                 int statoff, _Float16* __restrict__ y) {
    const size_t idx = ((size_t)blockIdx.x * blockDim.x + threadIdx.x) * 4;
    const size_t n = idx / ((size_t)C_ * L_);
    const float inv = 1.0f / ((float)C_ * (float)L_);
    const float mean = stats[(statoff + n) * 2 + 0] * inv;
    const float var  = stats[(statoff + n) * 2 + 1] * inv - mean * mean;
    const float rstd = rsqrtf(var + EPS_);
    const int c = (int)((idx / L_) % C_);
    const float sw = rstd * w[c], sb = bb[c];
#pragma unroll
    for (int i = 0; i < 4; ++i) {
        float v = (x[idx + i] - mean) * sw + sb;
        y[idx + i] = (_Float16)v;
    }
}

__global__ void cast_f16_kernel(const float* __restrict__ x, _Float16* __restrict__ y, int nElem) {
    int i = blockIdx.x * blockDim.x + threadIdx.x;
    if (i < nElem) y[i] = (_Float16)x[i];
}

// ---------------------------------------------------------------- WMMA GEMM
// out[n] = W[M,K] * X[n][K,L] + bias (+ resid), K=512, L=2048.
// 256 threads (8 waves), 128x128 output tile, 32-wide K steps.
template<bool F16OUT, bool RESID>
__global__ void gemm_wmma_kernel(const _Float16* __restrict__ W, const _Float16* __restrict__ X,
                                 const float* __restrict__ bias, const float* __restrict__ resid,
                                 void* __restrict__ outp, int M) {
    constexpr int K = C_;
    constexpr int L = L_;
    __shared__ _Float16 As[128][40];   // [m][k], stride 80B (16B multiple)
    __shared__ _Float16 BsT[128][40];  // [col][k] transposed

    const int n  = blockIdx.z;
    const int bm = blockIdx.y * 128;
    const int bl = blockIdx.x * 128;
    const _Float16* Xn = X + (size_t)n * K * L;
    const int tid = threadIdx.x;
    const int lane = tid & 31, w = tid >> 5;
    const int half = lane >> 4, l15 = lane & 15;
    const int waveM = w >> 1, waveN = w & 1;

    v8f acc[2][4];
#pragma unroll
    for (int rt = 0; rt < 2; ++rt)
#pragma unroll
        for (int ct = 0; ct < 4; ++ct)
#pragma unroll
            for (int r = 0; r < 8; ++r) acc[rt][ct][r] = 0.0f;

    for (int kt = 0; kt < K / 32; ++kt) {
        __syncthreads();
        {   // A tile: 128 rows x 32 k; thread: row=tid>>1, 16 f16 segment
            const int row = tid >> 1, seg = tid & 1;
            const _Float16* srcp = W + (size_t)(bm + row) * K + kt * 32 + seg * 16;
            const uint4* src = reinterpret_cast<const uint4*>(srcp);
            uint4 q0 = src[0], q1 = src[1];
            *reinterpret_cast<uint4*>(&As[row][seg * 16 + 0]) = q0;
            *reinterpret_cast<uint4*>(&As[row][seg * 16 + 8]) = q1;
            if (kt + 1 < K / 32) __builtin_prefetch(srcp + 32, 0, 1);
        }
        {   // B tile transposed: 32 k rows x 128 cols; thread: krow=tid>>3, 16 cols
            const int krow = tid >> 3, seg = tid & 7;
            const _Float16* src = Xn + (size_t)(kt * 32 + krow) * L + bl + seg * 16;
            H8 t0, t1;
            t0.u = *reinterpret_cast<const uint4*>(src);
            t1.u = *reinterpret_cast<const uint4*>(src + 8);
#pragma unroll
            for (int j = 0; j < 8; ++j) {
                BsT[seg * 16 + j][krow]     = t0.h[j];
                BsT[seg * 16 + 8 + j][krow] = t1.h[j];
            }
            if (kt + 1 < K / 32) __builtin_prefetch(src + (size_t)32 * L, 0, 1);
        }
        __syncthreads();

        v16h af[2], bf[4];
#pragma unroll
        for (int rt = 0; rt < 2; ++rt)
            af[rt] = ld_a32(&As[32 * waveM + 16 * rt + l15][0], half);
#pragma unroll
        for (int ct = 0; ct < 4; ++ct)
            bf[ct] = ld_b16(&BsT[64 * waveN + 16 * ct + l15][0], half);
#pragma unroll
        for (int rt = 0; rt < 2; ++rt)
#pragma unroll
            for (int ct = 0; ct < 4; ++ct)
                acc[rt][ct] = wmma16(af[rt], bf[ct], acc[rt][ct]);
    }

    // epilogue: C layout row = r + 8*half, col = lane&15
#pragma unroll
    for (int rt = 0; rt < 2; ++rt)
#pragma unroll
        for (int ct = 0; ct < 4; ++ct)
#pragma unroll
            for (int r = 0; r < 8; ++r) {
                const int row = bm + 32 * waveM + 16 * rt + r + 8 * half;
                const int col = bl + 64 * waveN + 16 * ct + l15;
                float v = acc[rt][ct][r] + bias[row];
                const size_t oidx = (size_t)n * M * L + (size_t)row * L + col;
                if (RESID) v += resid[oidx];
                if (F16OUT) ((_Float16*)outp)[oidx] = (_Float16)v;
                else        ((float*)outp)[oidx]    = v;
            }
}

// ---------------------------------------------------------------- flash attention
// Q from QKVq, K/V from QKVkv (both [N][3C][L] f16). out [N][C][L] f16.
// Block = (n, h, 128 queries), 8 waves x 16-query tiles, keys streamed 64 at a time.
// Softmax in exp2 domain (log2e folded into Q scale); row max via permlane16
// butterfly (VALU only); row sums via WMMA against a ones matrix.
__global__ void attn_kernel(const _Float16* __restrict__ QKVq,
                            const _Float16* __restrict__ QKVkv,
                            _Float16* __restrict__ out) {
    constexpr int L = L_;
    __shared__ _Float16 KsT[64][72];    // [key][d]  (transposed for B-frags)
    __shared__ _Float16 Vs[64][72];     // [dv][key] (natural = memory order)
    __shared__ _Float16 Ps[8][16][64];  // per-wave P: [q][key]

    const int qb = blockIdx.x * 128;
    const int h = blockIdx.y;
    const int n = blockIdx.z;
    const int tid = threadIdx.x, lane = tid & 31, w = tid >> 5;
    const int half = lane >> 4, l15 = lane & 15;

    const size_t base = (size_t)n * (3 * C_) * L;
    const _Float16* Qp = QKVq  + base + (size_t)(0 * C_ + h * DH_) * L;
    const _Float16* Kp = QKVkv + base + (size_t)(1 * C_ + h * DH_) * L;
    const _Float16* Vp = QKVkv + base + (size_t)(2 * C_ + h * DH_) * L;

    // Q fragments (softmax scale * log2e folded in), cached for the whole key loop
    const float qscale = 0.125f * 1.44269504088896340736f;
    const int qrow = qb + w * 16 + l15;
    v16h qf[2];
#pragma unroll
    for (int f = 0; f < 2; ++f)
#pragma unroll
        for (int j = 0; j < 16; ++j) {
            const int d = (j & 7) + 8 * half + 16 * (j >> 3) + 32 * f;
            qf[f][j] = (_Float16)((float)Qp[(size_t)d * L + qrow] * qscale);
        }

    v16h onesf;
#pragma unroll
    for (int j = 0; j < 16; ++j) onesf[j] = (_Float16)1.0f;

    v8f o[4];
    float rmax[8], rsum[8];
#pragma unroll
    for (int t = 0; t < 4; ++t)
#pragma unroll
        for (int r = 0; r < 8; ++r) o[t][r] = 0.0f;
#pragma unroll
    for (int r = 0; r < 8; ++r) { rmax[r] = -3.0e38f; rsum[r] = 0.0f; }

    for (int kc = 0; kc < L; kc += 64) {
        __syncthreads();
        {   // stage K (transposed) and V (natural): thread covers (d=tid>>2, 16 keys)
            const int d = tid >> 2, c0 = (tid & 3) * 16;
            const _Float16* ksrc = Kp + (size_t)d * L + kc + c0;
            const _Float16* vsrc = Vp + (size_t)d * L + kc + c0;
            H8 tk0, tk1;
            tk0.u = *reinterpret_cast<const uint4*>(ksrc);
            tk1.u = *reinterpret_cast<const uint4*>(ksrc + 8);
#pragma unroll
            for (int j = 0; j < 8; ++j) {
                KsT[c0 + j][d]     = tk0.h[j];
                KsT[c0 + 8 + j][d] = tk1.h[j];
            }
            *reinterpret_cast<uint4*>(&Vs[d][c0])     = *reinterpret_cast<const uint4*>(vsrc);
            *reinterpret_cast<uint4*>(&Vs[d][c0 + 8]) = *reinterpret_cast<const uint4*>(vsrc + 8);
            if (kc + 64 < L) {
                __builtin_prefetch(ksrc + 64, 0, 1);
                __builtin_prefetch(vsrc + 64, 0, 1);
            }
        }
        __syncthreads();

        // S = (Q*scale) K^T  -> four 16x16 tiles (keys 0-15,16-31,32-47,48-63)
        v8f s[4];
#pragma unroll
        for (int ct = 0; ct < 4; ++ct)
#pragma unroll
            for (int r = 0; r < 8; ++r) s[ct][r] = 0.0f;
#pragma unroll
        for (int p = 0; p < 2; ++p)
#pragma unroll
            for (int ct = 0; ct < 4; ++ct) {
                v16h bk = ld_b16(&KsT[16 * ct + l15][32 * p], half);
                s[ct] = wmma16(qf[p], bk, s[ct]);
            }

        // online softmax: row (r + 8*half) lives across the 16-lane half-group.
#pragma unroll
        for (int r = 0; r < 8; ++r) {
            const float m = rowmax16(fmaxf(fmaxf(s[0][r], s[1][r]), fmaxf(s[2][r], s[3][r])));
            const float nm = fmaxf(rmax[r], m);
            const float fac = exp2f(rmax[r] - nm);
            rmax[r] = nm;
            rsum[r] *= fac;
            o[0][r] *= fac; o[1][r] *= fac; o[2][r] *= fac; o[3][r] *= fac;
#pragma unroll
            for (int ct = 0; ct < 4; ++ct) {
                const float p = exp2f(s[ct][r] - nm);
                Ps[w][r + 8 * half][16 * ct + l15] = (_Float16)p;
            }
        }
        // Per-wave LDS slab: same-wave DS ordering + compiler dscnt waits order
        // the Ps stores against the A-fragment loads below (no block barrier needed).
        v16h pf0 = ld_a32(&Ps[w][l15][0], half);
        v16h pf1 = ld_a32(&Ps[w][l15][32], half);

        // row sums: P(16x64) x ones(64x16); every lane's C element = its row's sum
        v8f st;
#pragma unroll
        for (int r = 0; r < 8; ++r) st[r] = 0.0f;
        st = wmma16(pf0, onesf, st);
        st = wmma16(pf1, onesf, st);
#pragma unroll
        for (int r = 0; r < 8; ++r) rsum[r] += st[r];

        // O += P V : B = V (64 keys x 16 dv) per dv tile, two 32-key halves
#pragma unroll
        for (int t = 0; t < 4; ++t) {
            v16h vf0 = ld_b16(&Vs[16 * t + l15][0], half);
            v16h vf1 = ld_b16(&Vs[16 * t + l15][32], half);
            o[t] = wmma16(pf0, vf0, o[t]);
            o[t] = wmma16(pf1, vf1, o[t]);
        }
    }

    // out[n][h*64 + dv][q] = o / rsum
    _Float16* outn = out + (size_t)n * C_ * L;
    float rinv[8];
#pragma unroll
    for (int r = 0; r < 8; ++r) rinv[r] = fast_rcp(rsum[r]);
#pragma unroll
    for (int t = 0; t < 4; ++t)
#pragma unroll
        for (int r = 0; r < 8; ++r) {
            const int dv = h * DH_ + 16 * t + l15;
            const int q = qb + w * 16 + r + 8 * half;
            outn[(size_t)dv * L + q] = (_Float16)(o[t][r] * rinv[r]);
        }
}

// ---------------------------------------------------------------- launch
extern "C" void kernel_launch(void* const* d_in, const int* in_sizes, int n_in,
                              void* d_out, int out_size, void* d_ws, size_t ws_size,
                              hipStream_t stream) {
    (void)in_sizes; (void)n_in; (void)out_size; (void)ws_size;
    const float* a       = (const float*)d_in[0];
    const float* b       = (const float*)d_in[1];
    const float* gn_a_w  = (const float*)d_in[2];
    const float* gn_a_b  = (const float*)d_in[3];
    const float* gn_b_w  = (const float*)d_in[4];
    const float* gn_b_b  = (const float*)d_in[5];
    const float* qkv_a_w = (const float*)d_in[6];
    const float* qkv_a_b = (const float*)d_in[7];
    const float* qkv_b_w = (const float*)d_in[8];
    const float* qkv_b_b = (const float*)d_in[9];
    const float* out_a_w = (const float*)d_in[10];
    const float* out_a_b = (const float*)d_in[11];
    const float* out_b_w = (const float*)d_in[12];
    const float* out_b_b = (const float*)d_in[13];

    char* ws = (char*)d_ws;
    size_t off = 0;
    auto alloc = [&](size_t bytes) -> void* {
        void* p = ws + off;
        off += (bytes + 255) & ~(size_t)255;
        return p;
    };
    const size_t NCL = (size_t)N_ * C_ * L_;
    float*     stats = (float*)alloc(16 * sizeof(float));
    _Float16*  XnA   = (_Float16*)alloc(NCL * 2);
    _Float16*  XnB   = (_Float16*)alloc(NCL * 2);
    _Float16*  WqA   = (_Float16*)alloc((size_t)3 * C_ * C_ * 2);
    _Float16*  WqB   = (_Float16*)alloc((size_t)3 * C_ * C_ * 2);
    _Float16*  WoA   = (_Float16*)alloc((size_t)C_ * C_ * 2);
    _Float16*  WoB   = (_Float16*)alloc((size_t)C_ * C_ * 2);
    _Float16*  QKVa  = (_Float16*)alloc((size_t)N_ * 3 * C_ * L_ * 2);
    _Float16*  QKVb  = (_Float16*)alloc((size_t)N_ * 3 * C_ * L_ * 2);
    _Float16*  AtA   = (_Float16*)alloc(NCL * 2);
    _Float16*  AtB   = (_Float16*)alloc(NCL * 2);

    zero_stats_kernel<<<1, 32, 0, stream>>>(stats);
    stats_kernel<<<dim3(256, N_, 2), 256, 0, stream>>>(a, b, stats);
    normalize_kernel<<<dim3((unsigned)(NCL / 1024)), 256, 0, stream>>>(a, gn_a_w, gn_a_b, stats, 0,  XnA);
    normalize_kernel<<<dim3((unsigned)(NCL / 1024)), 256, 0, stream>>>(b, gn_b_w, gn_b_b, stats, N_, XnB);

    cast_f16_kernel<<<dim3((3 * C_ * C_ + 255) / 256), 256, 0, stream>>>(qkv_a_w, WqA, 3 * C_ * C_);
    cast_f16_kernel<<<dim3((3 * C_ * C_ + 255) / 256), 256, 0, stream>>>(qkv_b_w, WqB, 3 * C_ * C_);
    cast_f16_kernel<<<dim3((C_ * C_ + 255) / 256), 256, 0, stream>>>(out_a_w, WoA, C_ * C_);
    cast_f16_kernel<<<dim3((C_ * C_ + 255) / 256), 256, 0, stream>>>(out_b_w, WoB, C_ * C_);

    gemm_wmma_kernel<true, false><<<dim3(L_ / 128, 3 * C_ / 128, N_), 256, 0, stream>>>(
        WqA, XnA, qkv_a_b, nullptr, QKVa, 3 * C_);
    gemm_wmma_kernel<true, false><<<dim3(L_ / 128, 3 * C_ / 128, N_), 256, 0, stream>>>(
        WqB, XnB, qkv_b_b, nullptr, QKVb, 3 * C_);

    attn_kernel<<<dim3(L_ / 128, H_, N_), 256, 0, stream>>>(QKVa, QKVb, AtA);  // a attends b
    attn_kernel<<<dim3(L_ / 128, H_, N_), 256, 0, stream>>>(QKVb, QKVa, AtB);  // b attends a

    float* outA = (float*)d_out;
    float* outB = outA + NCL;
    gemm_wmma_kernel<false, true><<<dim3(L_ / 128, C_ / 128, N_), 256, 0, stream>>>(
        WoA, AtA, out_a_b, a, outA, C_);
    gemm_wmma_kernel<false, true><<<dim3(L_ / 128, C_ / 128, N_), 256, 0, stream>>>(
        WoB, AtB, out_b_b, b, outB, C_);
}
